// RNN_24781961298493
// MI455X (gfx1250) — matrix operand order repaired and checked by reading, not compiled, not run on hardware
//
#include <hip/hip_runtime.h>

// ---------------------------------------------------------------------------
// Fused embedding + input-projection + LSTM(T=64) + FC kernel for gfx1250.
//
// Design (MI455X reasoning):
//  * Never materialize x_proj[B,T,256] (1 GB of HBM -> fused away).
//  * VOCAB=25: precompute xproj_table[25->32, 256] = emb@W_ih^T + b_ih + b_hh.
//    Per step, gate init = OneHot(16x32) @ xproj_table  -- an extra WMMA,
//    with the one-hot A operand built in registers from the token.
//    The 16 xproj B-fragments are step-invariant -> held in 128 VGPRs for
//    the whole t-loop (cuts per-step LDS traffic 48KB -> 32KB).
//  * Recurrence gates = init + h(16x64 bf16) @ W_hh^T via 2x
//    v_wmma_f32_16x16x32_bf16 per 16x16 gate tile (f32 accumulate).
//  * W_hh fragments pre-packed into LDS in the ISA B-matrix lane layout
//    (VGPR v, lane l<16: K=2(v&3)+16(v>>2) and +1, N=l ; lanes>=16: K+=8).
//    Per-lane 8 contiguous dwords, lane stride padded to 12 dwords
//    (48 B, 16B-aligned) -> 2-way-max bank pattern for ds_load_b128.
//  * h state relaid C-layout -> A-layout through per-wave padded LDS
//    scratch (16 x 72 bf16; row stride 144 B gives conflict-free b128
//    loads). Same-wave producer/consumer only: DScnt ordering, no barrier.
//  * One wave = 16 batch rows; block = 4 waves; grid = 256 blocks
//    (1024 waves total; t-loop latency hidden by deep intra-wave
//    pipelining -- the scheduler keeps ~30 DS loads in flight).
//  * Native v_tanh_f32 (confirmed in round 2): 1 trans per activation;
//    sigmoid(x) = 0.5*tanh(0.5x)+0.5.
// ---------------------------------------------------------------------------

#define VOCABSZ 25
#define EMBSZ   32
#define HIDSZ   64
#define TSTEPS  64
#define BATCH   16384
#define OUTSZ   3

#define WAVES          4
#define BLOCK          (WAVES * 32)
#define ROWS_PER_WAVE  16
#define ROWS_PER_BLOCK (WAVES * ROWS_PER_WAVE)
#define HPAD           72          // padded bf16 row stride for h scratch
#define FRDW           384         // dwords per fragment slot (12/lane * 32)

typedef __attribute__((ext_vector_type(16))) __bf16    v16bf;
typedef __attribute__((ext_vector_type(8)))  float     v8f;
typedef __attribute__((ext_vector_type(4)))  unsigned  u32x4;

union FragU { v16bf v; u32x4 q[2]; unsigned u[8]; };

__device__ __forceinline__ unsigned short f32_to_bf16(float x) {
  unsigned u = __builtin_bit_cast(unsigned, x);
  u += 0x7FFFu + ((u >> 16) & 1u);                  // round-to-nearest-even
  return (unsigned short)(u >> 16);
}
__device__ __forceinline__ float bf16_to_f32(unsigned short b) {
  return __builtin_bit_cast(float, ((unsigned)b) << 16);
}
__device__ __forceinline__ unsigned pack2(unsigned short lo, unsigned short hi) {
  return (unsigned)lo | ((unsigned)hi << 16);
}

#if __has_builtin(__builtin_amdgcn_tanhf)
// Native v_tanh_f32 (CDNA5 TRANS op): 1 trans per activation.
__device__ __forceinline__ float fast_tanh(float x) {
  return __builtin_amdgcn_tanhf(x);
}
__device__ __forceinline__ float fast_sigmoid(float x) {
  return 0.5f * __builtin_amdgcn_tanhf(0.5f * x) + 0.5f;
}
#else
__device__ __forceinline__ float fast_tanh(float x) {
  float e = __builtin_amdgcn_exp2f(2.8853900817779268f * x);
  return 1.0f - 2.0f * __builtin_amdgcn_rcpf(e + 1.0f);
}
__device__ __forceinline__ float fast_sigmoid(float x) {
  float e = __builtin_amdgcn_exp2f(-1.4426950408889634f * x);
  return __builtin_amdgcn_rcpf(1.0f + e);
}
#endif

__device__ __forceinline__ v16bf ld_frag(const unsigned* p) {   // 2x ds_load_b128
  FragU f;
  f.q[0] = *(const u32x4*)(p);
  f.q[1] = *(const u32x4*)(p + 4);
  return f.v;
}

__global__ __launch_bounds__(BLOCK) void lstm_fused_kernel(
    const int*   __restrict__ msg,     // [B, T]
    const float* __restrict__ emb,     // [25, 32]
    const float* __restrict__ W_ih,    // [256, 32]
    const float* __restrict__ W_hh,    // [256, 64]
    const float* __restrict__ b_ih,    // [256]
    const float* __restrict__ b_hh,    // [256]
    const float* __restrict__ fc_w,    // [3, 64]
    const float* __restrict__ fc_b,    // [3]
    float*       __restrict__ out)     // [B, 3]
{
  __shared__ __align__(16) unsigned       s_wfrag[32 * FRDW];        // 48 KB
  __shared__ __align__(16) unsigned       s_xfrag[16 * FRDW];        // 24 KB
  __shared__ __align__(16) unsigned short s_h[WAVES * 16 * HPAD];    //  9 KB

  const int tid = threadIdx.x;

  // ---- Phase 0: pack W_hh^T into bf16 B-matrix fragments ------------------
  // Fragment f = tile n (f>>1) x K-half (f&1). Dword (v, l):
  //   K = kb + 2(v&3) + 16(v>>2) + (l<16 ? 0 : 8), N(gate col) = n*16 + (l&15)
  for (int idx = tid; idx < 32 * 256; idx += BLOCK) {
    int f = idx >> 8, d = idx & 255;
    int l = d & 31, v = d >> 5;
    int n = f >> 1, kb = (f & 1) * 32;
    int g  = n * 16 + (l & 15);
    int k0 = kb + 2 * (v & 3) + ((v >> 2) << 4) + ((l < 16) ? 0 : 8);
    float w0 = W_hh[g * HIDSZ + k0];
    float w1 = W_hh[g * HIDSZ + k0 + 1];
    s_wfrag[f * FRDW + l * 12 + v] = pack2(f32_to_bf16(w0), f32_to_bf16(w1));
  }
  // ---- Phase 0b: build xproj_table rows on the fly, pack as B fragments ---
  // xproj[tok][g] = dot(emb[tok], W_ih[g]) + b_ih[g] + b_hh[g]; rows >=25 are 0.
  for (int idx = tid; idx < 16 * 256; idx += BLOCK) {
    int n = idx >> 8, d = idx & 255;
    int l = d & 31, v = d >> 5;
    int g  = n * 16 + (l & 15);
    int k0 = 2 * (v & 3) + ((v >> 2) << 4) + ((l < 16) ? 0 : 8);          // vocab index (K dim)
    float bias = b_ih[g] + b_hh[g];
    float x0 = 0.f, x1 = 0.f;
    if (k0 < VOCABSZ) {
      for (int e = 0; e < EMBSZ; ++e) x0 += emb[k0 * EMBSZ + e] * W_ih[g * EMBSZ + e];
      x0 += bias;
    }
    if (k0 + 1 < VOCABSZ) {
      for (int e = 0; e < EMBSZ; ++e) x1 += emb[(k0 + 1) * EMBSZ + e] * W_ih[g * EMBSZ + e];
      x1 += bias;
    }
    s_xfrag[n * FRDW + l * 12 + v] = pack2(f32_to_bf16(x0), f32_to_bf16(x1));
  }
  __syncthreads();

  // ---- Phase 1: LSTM recurrence, one wave per 16 batch rows ---------------
  const int lane = tid & 31;
  const int wave = tid >> 5;
  const int row_base = blockIdx.x * ROWS_PER_BLOCK + wave * ROWS_PER_WAVE;
  const int my_row   = row_base + (lane & 15);     // lanes l and l+16 share a row
  unsigned short* hscr = &s_h[wave * 16 * HPAD];

  const unsigned* wbase = &s_wfrag[lane * 12];
  const unsigned* xbase = &s_xfrag[lane * 12];
  const int koffA = (lane < 16) ? 0 : 8;           // A-layout K offset for this half-wave

  // xproj fragments are step-invariant: pin all 16 in VGPRs for the t-loop.
  v16bf xf[16];
  #pragma unroll
  for (int n = 0; n < 16; ++n) xf[n] = ld_frag(xbase + n * FRDW);

  v8f c[4];
  #pragma unroll
  for (int j = 0; j < 4; ++j) c[j] = (v8f){0.f, 0.f, 0.f, 0.f, 0.f, 0.f, 0.f, 0.f};

  FragU hz; hz.q[0] = (u32x4)(0u); hz.q[1] = (u32x4)(0u);
  v16bf hA0 = hz.v, hA1 = hz.v;                    // h0 = 0

  int tok = msg[my_row * TSTEPS];                  // token for t = 0

  for (int t = 0; t < TSTEPS; ++t) {
    // Prefetch next token early so the (strided, L2-resident) global load
    // overlaps this step's math instead of stalling the next one-hot build.
    int tok_next = (t + 1 < TSTEPS) ? msg[my_row * TSTEPS + t + 1] : 0;

    // Build one-hot A operand (16x32 bf16): row M = lane&15 has 1.0 at K = tok.
    // kA is always even -> one compare + one select per dword.
    const unsigned onebits = 0x3F80u << ((tok & 1) << 4);
    const int      toke    = tok & ~1;
    FragU oh;
    #pragma unroll
    for (int v = 0; v < 8; ++v) {
      int kA = 2 * (v & 3) + ((v >> 2) << 4) + koffA;
      oh.u[v] = (toke == kA) ? onebits : 0u;
    }

    #pragma unroll
    for (int j = 0; j < 4; ++j) {                  // hidden-column group (16 cols)
      v8f acc[4];
      #pragma unroll
      for (int q = 0; q < 4; ++q) {                // gate: i, f, g, o
        const int n = q * 4 + j;                   // gate tile index 0..15
        v8f z = (v8f){0.f, 0.f, 0.f, 0.f, 0.f, 0.f, 0.f, 0.f};
        v8f a = __builtin_amdgcn_wmma_f32_16x16x32_bf16(
                    false, oh.v, false, xf[n], (short)0, z, false, false);
        v16bf bw0 = ld_frag(wbase + (2 * n) * FRDW);
        a = __builtin_amdgcn_wmma_f32_16x16x32_bf16(
                    false, hA0, false, bw0, (short)0, a, false, false);
        v16bf bw1 = ld_frag(wbase + (2 * n + 1) * FRDW);
        a = __builtin_amdgcn_wmma_f32_16x16x32_bf16(
                    false, hA1, false, bw1, (short)0, a, false, false);
        acc[q] = a;
      }
      // Pointwise LSTM cell update in f32, C-layout.
      #pragma unroll
      for (int e = 0; e < 8; ++e) {
        float ig = fast_sigmoid(acc[0][e]);
        float fg = fast_sigmoid(acc[1][e]);
        float gg = fast_tanh   (acc[2][e]);
        float og = fast_sigmoid(acc[3][e]);
        float cn = fg * c[j][e] + ig * gg;
        c[j][e]  = cn;
        float hv = og * fast_tanh(cn);
        int r   = e + ((lane < 16) ? 0 : 8);       // C layout: rows m / m+8
        int col = j * 16 + (lane & 15);
        hscr[r * HPAD + col] = f32_to_bf16(hv);    // ds_store_b16
      }
    }

    // Relayout h: read back as two 16x32 bf16 A fragments (4x ds_load_b128).
    const unsigned short* hrow = hscr + (lane & 15) * HPAD;
    FragU f0, f1;
    f0.q[0] = *(const u32x4*)(const void*)(hrow + koffA);
    f0.q[1] = *(const u32x4*)(const void*)(hrow + koffA + 16);
    f1.q[0] = *(const u32x4*)(const void*)(hrow + koffA + 32);
    f1.q[1] = *(const u32x4*)(const void*)(hrow + koffA + 48);
    hA0 = f0.v;
    hA1 = f1.v;

    tok = tok_next;
  }

  // ---- Phase 2: final FC, h_T[16,64] @ fc_w^T[64,3] + fc_b ----------------
  if (lane < 16) {
    const unsigned short* hr = hscr + lane * HPAD;
    float o0 = fc_b[0], o1 = fc_b[1], o2 = fc_b[2];
    for (int n2 = 0; n2 < HIDSZ; ++n2) {
      float hv = bf16_to_f32(hr[n2]);
      o0 += hv * fc_w[0 * HIDSZ + n2];
      o1 += hv * fc_w[1 * HIDSZ + n2];
      o2 += hv * fc_w[2 * HIDSZ + n2];
    }
    float* op = out + (size_t)(row_base + lane) * OUTSZ;
    op[0] = o0; op[1] = o1; op[2] = o2;
  }
}

extern "C" void kernel_launch(void* const* d_in, const int* in_sizes, int n_in,
                              void* d_out, int out_size, void* d_ws, size_t ws_size,
                              hipStream_t stream) {
  (void)in_sizes; (void)n_in; (void)out_size; (void)d_ws; (void)ws_size;
  const int*   msg  = (const int*)  d_in[0];
  const float* emb  = (const float*)d_in[1];
  const float* W_ih = (const float*)d_in[2];
  const float* W_hh = (const float*)d_in[3];
  const float* b_ih = (const float*)d_in[4];
  const float* b_hh = (const float*)d_in[5];
  const float* fc_w = (const float*)d_in[6];
  const float* fc_b = (const float*)d_in[7];
  float* out = (float*)d_out;

  dim3 grid(BATCH / ROWS_PER_BLOCK);   // 256 workgroups
  dim3 block(BLOCK);                   // 128 threads = 4 wave32
  lstm_fused_kernel<<<grid, block, 0, stream>>>(msg, emb, W_ih, W_hh,
                                                b_ih, b_hh, fc_w, fc_b, out);
}
